// NWFunctional_79697413144887
// MI455X (gfx1250) — matrix operand order repaired and checked by reading, not compiled, not run on hardware
//
#include <hip/hip_runtime.h>
#include <hip/hip_bf16.h>

// Nadaraya-Watson kernel regression, fused flash-attention style, for gfx1250.
//   K = exp(-0.5*(||q||^2 + ||k||^2 - 2 q.k)/l^2);  out = (K @ Y) / (rowsum(K)+eps)
// GEMM1 (q.k^T): bf16 hi/lo split (3 wmma per k-step) for near-f32 accuracy.
//   Xtrain's bf16 hi/lo split and row norms are precomputed ONCE into d_ws so the
//   hot loop issues pure b128 loads + wmma (no per-iteration convert VALU).
// GEMM2 (K @ Y): plain bf16, Y pre-transposed+converted into d_ws.
// Grid: 128 blocks x 4 waves (16 test rows per wave) for full-WGP coverage;
// the 36 KB per-iteration B/Y tile is shared by all waves via WGP$ hits.
// Workspace layout (needs ~36.2 MB):
//   [0,4MB)    Ybf  : bf16 [64][32768]   (Ytrain transposed)
//   [4,20MB)   Xhi  : bf16 [32768][256]  (Xtrain hi)
//   [20,36MB)  Xlo  : bf16 [32768][256]  (Xtrain residual)
//   [36MB,+128KB) Ksq : f32 [32768]      (train row norms)

typedef __attribute__((ext_vector_type(16))) __bf16 v16bf;
typedef __attribute__((ext_vector_type(8)))  float  v8f;

#define NTRAIN 32768
#define NTEST  8192
#define DIM    256
#define DY     64

// ---- prep: Ytrain [32768 x 64] f32 -> Ybf [64 x 32768] bf16 ----
__global__ void ytrans_kernel(const float* __restrict__ Y, __bf16* __restrict__ Ybf) {
    int idx = blockIdx.x * 256 + threadIdx.x;       // over 64*32768 outputs
    int c = idx >> 15;                              // / 32768
    int n = idx & (NTRAIN - 1);
    Ybf[(size_t)c * NTRAIN + n] = (__bf16)Y[(size_t)n * DY + c];
}

// ---- prep: Xtrain f32 -> bf16 hi + residual lo (row-major, same layout) ----
__global__ void xsplit_kernel(const float* __restrict__ X,
                              __bf16* __restrict__ Xhi, __bf16* __restrict__ Xlo) {
    int idx = (blockIdx.x * 256 + threadIdx.x) * 4; // over 32768*256 elements
    float4 f = *(const float4*)(X + idx);
    float x[4] = {f.x, f.y, f.z, f.w};
#pragma unroll
    for (int j = 0; j < 4; ++j) {
        __bf16 h = (__bf16)x[j];
        Xhi[idx + j] = h;
        Xlo[idx + j] = (__bf16)(x[j] - (float)h);
    }
}

// ---- prep: train row norms ||xtrain_n||^2 ----
__global__ void ksq_kernel(const float* __restrict__ X, float* __restrict__ Ksq) {
    int n = blockIdx.x * 256 + threadIdx.x;
    const float* p = X + (size_t)n * DIM;
    float s = 0.f;
#pragma unroll 8
    for (int i = 0; i < DIM; i += 4) {
        float4 f = *(const float4*)(p + i);
        s += f.x * f.x + f.y * f.y + f.z * f.z + f.w * f.w;
    }
    Ksq[n] = s;
}

// load 8 consecutive f32, split into bf16 hi/lo at vector offset `off`,
// accumulate sum of squares (used only in the once-per-wave Xtest prologue).
__device__ __forceinline__ void cvt8_split(const float* __restrict__ p,
                                           v16bf& hi, v16bf& lo, int off, float& ss) {
    float w[8];
    *(float4*)(w + 0) = *(const float4*)(p + 0);
    *(float4*)(w + 4) = *(const float4*)(p + 4);
#pragma unroll
    for (int i = 0; i < 8; ++i) {
        float t = w[i];
        __bf16 h = (__bf16)t;
        hi[off + i] = h;
        lo[off + i] = (__bf16)(t - (float)h);
        ss += t * t;
    }
}

__global__ __launch_bounds__(128, 1)
void nw_kernel(const float* __restrict__ Xtest, const float* __restrict__ logl,
               const __bf16* __restrict__ Xhi, const __bf16* __restrict__ Xlo,
               const float* __restrict__ Ksq, const __bf16* __restrict__ Ybf,
               float* __restrict__ out) {
    __shared__ float ldsK[4 * 16 * 32];             // per-wave 16x32 K-tile scratch

    const int lane   = threadIdx.x & 31;
    const int wave   = threadIdx.x >> 5;            // 0..3
    const int lane16 = lane & 15;
    const int half   = lane >> 4;
    const int koffA  = half * 16;                   // A-operand K range per half-wave
    const int koffB  = half * 16;                   // B-operand K range per half-wave
    const int half8  = half * 8;                    // C/D row group per half-wave
    float* lk = &ldsK[wave * 512];

    const int m0 = blockIdx.x * 64 + wave * 16;     // this wave's 16 test rows
    const float inv = 0.5f * __expf(-2.0f * logl[0]);   // 0.5 / l^2

    // ---- prologue: load A strip (16 test rows x 256 feats) as bf16 hi/lo regs ----
    v16bf Ahi[8], Alo[8];
    float qp = 0.f;
    const float* xrow = Xtest + (size_t)(m0 + lane16) * DIM;
#pragma unroll
    for (int ks = 0; ks < 8; ++ks) {
        const float* p = xrow + ks * 32 + koffA;
        cvt8_split(p,      Ahi[ks], Alo[ks], 0, qp);
        cvt8_split(p + 8,  Ahi[ks], Alo[ks], 8, qp);
    }
    float qsq = qp + __shfl_xor(qp, 16);            // full ||q||^2 for row m0+lane16
    float qsqM[8];
#pragma unroll
    for (int v = 0; v < 8; ++v) qsqM[v] = __shfl(qsq, v + half8);

    v8f accN[4];                                    // numer: 16 rows x 64 cols
#pragma unroll
    for (int cc = 0; cc < 4; ++cc) accN[cc] = v8f{};
    float dsum[8] = {0.f, 0.f, 0.f, 0.f, 0.f, 0.f, 0.f, 0.f};

    for (int n0 = 0; n0 < NTRAIN; n0 += 32) {
        const float ksq0 = Ksq[n0 + lane16];        // tile0 column norm
        const float ksq1 = Ksq[n0 + 16 + lane16];   // tile1 column norm

        // ---- GEMM1: two 16x16 S tiles, K=256 in 8 steps, hi/lo split, pure loads ----
        v8f S0 = {}, S1 = {};
        const size_t r0 = (size_t)(n0 + lane16) * DIM + koffB;
        const size_t r1 = (size_t)(n0 + 16 + lane16) * DIM + koffB;
#pragma unroll
        for (int ks = 0; ks < 8; ++ks) {
            v16bf B0h = *(const v16bf*)(Xhi + r0 + ks * 32);
            v16bf B0l = *(const v16bf*)(Xlo + r0 + ks * 32);
            v16bf B1h = *(const v16bf*)(Xhi + r1 + ks * 32);
            v16bf B1l = *(const v16bf*)(Xlo + r1 + ks * 32);
            S0 = __builtin_amdgcn_wmma_f32_16x16x32_bf16(false, Ahi[ks], false, B0h,
                                                         (short)0, S0, false, false);
            S0 = __builtin_amdgcn_wmma_f32_16x16x32_bf16(false, Ahi[ks], false, B0l,
                                                         (short)0, S0, false, false);
            S0 = __builtin_amdgcn_wmma_f32_16x16x32_bf16(false, Alo[ks], false, B0h,
                                                         (short)0, S0, false, false);
            S1 = __builtin_amdgcn_wmma_f32_16x16x32_bf16(false, Ahi[ks], false, B1h,
                                                         (short)0, S1, false, false);
            S1 = __builtin_amdgcn_wmma_f32_16x16x32_bf16(false, Ahi[ks], false, B1l,
                                                         (short)0, S1, false, false);
            S1 = __builtin_amdgcn_wmma_f32_16x16x32_bf16(false, Alo[ks], false, B1h,
                                                         (short)0, S1, false, false);
        }

        // ---- exp, denom accumulate, stage K-tile row-major into LDS ----
#pragma unroll
        for (int v = 0; v < 8; ++v) {
            float sq0 = qsqM[v] + ksq0 - 2.0f * S0[v];
            float kv0 = __expf(-sq0 * inv);
            float sq1 = qsqM[v] + ksq1 - 2.0f * S1[v];
            float kv1 = __expf(-sq1 * inv);
            dsum[v] += kv0 + kv1;
            lk[(v + half8) * 32 + lane16]      = kv0;
            lk[(v + half8) * 32 + 16 + lane16] = kv1;
        }
        asm volatile("s_wait_dscnt 0x0" ::: "memory");  // store->load transpose fence

        // ---- reload as A-operand layout (16x32), convert to bf16 ----
        v16bf Ak;
        {
            const float* q = lk + lane16 * 32 + koffA;
            float w[16];
            *(float4*)(w + 0)  = *(const float4*)(q + 0);
            *(float4*)(w + 4)  = *(const float4*)(q + 4);
            *(float4*)(w + 8)  = *(const float4*)(q + 8);
            *(float4*)(w + 12) = *(const float4*)(q + 12);
#pragma unroll
            for (int i = 0; i < 16; ++i) Ak[i] = (__bf16)w[i];
        }

        // ---- GEMM2: numer += K(16x32) @ Y(32x64), 4 column chunks ----
#pragma unroll
        for (int cc = 0; cc < 4; ++cc) {
            const __bf16* yb = Ybf + (size_t)(cc * 16 + lane16) * NTRAIN + n0 + koffB;
            v16bf Bv = *(const v16bf*)yb;           // 16 contiguous bf16, 32B aligned
            accN[cc] = __builtin_amdgcn_wmma_f32_16x16x32_bf16(false, Ak, false, Bv,
                                                               (short)0, accN[cc], false, false);
        }
    }

    // ---- reduce denominator across the 16 columns held within each half-wave ----
#pragma unroll
    for (int m = 1; m < 16; m <<= 1) {
#pragma unroll
        for (int v = 0; v < 8; ++v) dsum[v] += __shfl_xor(dsum[v], m);
    }

    // ---- normalize and store: lane holds rows {half8+v}, col cc*16+lane16 ----
#pragma unroll
    for (int v = 0; v < 8; ++v) {
        float rden = 1.0f / (dsum[v] + 1e-8f);
        int   row  = m0 + half8 + v;
#pragma unroll
        for (int cc = 0; cc < 4; ++cc)
            out[(size_t)row * DY + cc * 16 + lane16] = accN[cc][v] * rden;
    }
}

extern "C" void kernel_launch(void* const* d_in, const int* in_sizes, int n_in,
                              void* d_out, int out_size, void* d_ws, size_t ws_size,
                              hipStream_t stream) {
    (void)in_sizes; (void)n_in; (void)out_size; (void)ws_size;
    const float* Ytrain = (const float*)d_in[0];
    const float* Xtrain = (const float*)d_in[1];
    const float* Xtest  = (const float*)d_in[2];
    const float* logl   = (const float*)d_in[3];
    float* out = (float*)d_out;

    char* ws = (char*)d_ws;                         // requires ws_size >= ~36.2 MB
    __bf16* Ybf = (__bf16*)(ws);
    __bf16* Xhi = (__bf16*)(ws + ((size_t)4 << 20));
    __bf16* Xlo = (__bf16*)(ws + ((size_t)20 << 20));
    float*  Ksq = (float*)(ws + ((size_t)36 << 20));

    ytrans_kernel<<<(NTRAIN * DY) / 256, 256, 0, stream>>>(Ytrain, Ybf);
    xsplit_kernel<<<(NTRAIN * DIM / 4) / 256, 256, 0, stream>>>(Xtrain, Xhi, Xlo);
    ksq_kernel<<<NTRAIN / 256, 256, 0, stream>>>(Xtrain, Ksq);
    nw_kernel<<<NTEST / 64, 128, 0, stream>>>(Xtest, logl, Xhi, Xlo, Ksq, Ybf, out);
}
